// RhythmMambaModel_15023795601430
// MI455X (gfx1250) — matrix operand order repaired and checked by reading, not compiled, not run on hardware
//
#include <hip/hip_runtime.h>
#include <hip/hip_bf16.h>
#include <math.h>

// ---------------- model constants ----------------
#define B_SZ    16
#define T_SZ    2048
#define DMODEL  160
#define DSTATE  64
#define DCONV   12
#define NBLK    8
#define CONVK   9
#define DINNER  320          // 2*DMODEL
#define DTRANK  10
#define NTOK    (B_SZ * T_SZ)      // 32768
#define XPN     144                // padded x_proj rows: [dt(10) pad(6) B(64) C(64)]
#define XP_BOFF 16                 // B starts at col 16 (16B-aligned byte offset 64)
#define XP_COFF 80                 // C starts at col 80

#define INW_CNT  (NBLK * 2 * DINNER * DMODEL)  // 819200
#define XPW_CNT  (NBLK * XPN * DINNER)         // 368640
#define OUTW_CNT (NBLK * DMODEL * DINNER)      // 409600

// ---------------- types ----------------
typedef __attribute__((ext_vector_type(16))) __bf16          v16bf;
typedef __attribute__((ext_vector_type(8)))  float           v8f;
typedef __attribute__((ext_vector_type(8)))  unsigned short  u16x8;
typedef __attribute__((ext_vector_type(16))) unsigned short  u16x16;

__device__ __forceinline__ unsigned short f32_to_bf16_rne(float f) {
  unsigned int u = __builtin_bit_cast(unsigned int, f);
  u += 0x7FFFu + ((u >> 16) & 1u);      // round-to-nearest-even
  return (unsigned short)(u >> 16);
}
__device__ __forceinline__ float fsigmoid(float x) {
  return 1.0f / (1.0f + __expf(-x));
}

// ---- CDNA5 async global->LDS copy (ASYNCcnt-tracked), see CDNA5 ISA ----
__device__ __forceinline__ void async_load_b128(unsigned lds_byte_off, const void* gaddr) {
  asm volatile("global_load_async_to_lds_b128 %0, %1, off"
               :: "v"(lds_byte_off), "v"(gaddr) : "memory");
}
__device__ __forceinline__ void wait_async0() {
  asm volatile("s_wait_asynccnt 0x0" ::: "memory");
}
__device__ __forceinline__ void wait_ds0() {
  asm volatile("s_wait_dscnt 0x0" ::: "memory");
}

// ---------------- conv stem + exact GELU ----------------
__global__ void stem_gelu_kernel(const float* __restrict__ x,
                                 const float* __restrict__ pre_w,
                                 const float* __restrict__ pre_b,
                                 float* __restrict__ h) {
  int idx = blockIdx.x * blockDim.x + threadIdx.x;
  if (idx >= NTOK * DMODEL) return;
  int c = idx % DMODEL;
  int tok = idx / DMODEL;
  int b = tok / T_SZ, t = tok % T_SZ;
  float acc = pre_b[c];
#pragma unroll
  for (int j = 0; j < CONVK; ++j) {
    int tt = t + j - CONVK / 2;
    float xv = (tt >= 0 && tt < T_SZ) ? x[b * T_SZ + tt] : 0.0f;
    acc += xv * pre_w[c * CONVK + j];
  }
  h[idx] = 0.5f * acc * (1.0f + erff(acc * 0.7071067811865476f));
}

// ---------------- LayerNorm -> bf16 (one wave per token, 160 = 5*32) ----------------
__global__ void layernorm_bf16_kernel(const float* __restrict__ h,
                                      const float* __restrict__ g,
                                      const float* __restrict__ beta,
                                      unsigned short* __restrict__ out_bf16) {
  int wid  = blockIdx.x * (blockDim.x >> 5) + (threadIdx.x >> 5);
  int lane = threadIdx.x & 31;
  if (wid >= NTOK) return;
  const float* row = h + (size_t)wid * DMODEL;
  float v[5], s = 0.f, s2 = 0.f;
#pragma unroll
  for (int i = 0; i < 5; ++i) {
    float xv = row[lane + 32 * i];
    v[i] = xv; s += xv; s2 += xv * xv;
  }
#pragma unroll
  for (int m = 16; m >= 1; m >>= 1) {
    s  += __shfl_xor(s,  m, 32);
    s2 += __shfl_xor(s2, m, 32);
  }
  float mean = s * (1.0f / DMODEL);
  float var  = s2 * (1.0f / DMODEL) - mean * mean;
  float rstd = rsqrtf(var + 1e-5f);
  unsigned short* orow = out_bf16 + (size_t)wid * DMODEL;
#pragma unroll
  for (int i = 0; i < 5; ++i) {
    int c = lane + 32 * i;
    orow[c] = f32_to_bf16_rne((v[i] - mean) * rstd * g[c] + beta[c]);
  }
}

// ---------------- fp32 -> bf16 weight conversion ----------------
// x_proj rows remapped to [dt(0..9) zero(10..15) B(16..79) C(80..143)] so the
// B/C block is 16B-aligned for async b128 copies in the scan.
__global__ void convert_weights_kernel(const float* __restrict__ in_w,
                                       const float* __restrict__ xproj_w,
                                       const float* __restrict__ out_w,
                                       unsigned short* __restrict__ wb_in,
                                       unsigned short* __restrict__ wb_xp,
                                       unsigned short* __restrict__ wb_out) {
  int idx = blockIdx.x * blockDim.x + threadIdx.x;
  if (idx < INW_CNT) { wb_in[idx] = f32_to_bf16_rne(in_w[idx]); return; }
  idx -= INW_CNT;
  if (idx < XPW_CNT) {
    int k   = idx % DINNER;
    int e   = (idx / DINNER) % XPN;
    int blk = idx / (DINNER * XPN);
    int src = (e < DTRANK) ? e : ((e >= XP_BOFF) ? (e - (XP_BOFF - DTRANK)) : -1);
    float v = (src >= 0)
                  ? xproj_w[((size_t)blk * (DTRANK + 2 * DSTATE) + src) * DINNER + k]
                  : 0.0f;
    wb_xp[idx] = f32_to_bf16_rne(v);
    return;
  }
  idx -= XPW_CNT;
  if (idx < OUTW_CNT) wb_out[idx] = f32_to_bf16_rne(out_w[idx]);
}

// ---------------- bf16 WMMA GEMM: C[M,N] = A[M,K] * W[N,K]^T (+ addsrc) ----------------
// One wave per 16x(16*NT) strip: A fragment reused across NT WMMAs per K-step.
// Requires M % 16 == 0, N % (16*NT) == 0, K % 32 == 0, rows 16B aligned.
template <int NT>
__global__ void gemm_bf16_wmma_kernel(const unsigned short* __restrict__ A, int lda,
                                      const unsigned short* __restrict__ W, int ldw,
                                      const float* __restrict__ addsrc,
                                      float* __restrict__ C, int ldc,
                                      int M, int N, int K) {
  int wid  = blockIdx.x * (blockDim.x >> 5) + (threadIdx.x >> 5);
  int lane = threadIdx.x & 31;
  int strips_n = N / (16 * NT);
  int tiles_m  = M >> 4;
  if (wid >= tiles_m * strips_n) return;
  int mt = wid / strips_n, ns = wid % strips_n;
  int m0 = mt << 4, n0 = ns * (16 * NT);

  int r15  = lane & 15;       // A row / B column within tile
  int ksel = lane >> 4;       // which K-half this lane holds
  // A fragment (16-bit, 16x32): lane<16 holds K 0..15, lane>=16 holds K 16..31
  const unsigned short* pa = A + (size_t)(m0 + r15) * lda + ksel * 16;
  // B fragment (32x16 col = W row): lane<16 holds K 0..15, lane>=16 holds K 16..31
  const unsigned short* pw = W + (size_t)(n0 + r15) * ldw + ksel * 16;

  v8f acc[NT] = {};
  for (int k = 0; k < K; k += 32) {
    u16x8 alo = *(const u16x8*)(pa + k);
    u16x8 ahi = *(const u16x8*)(pa + k + 8);
    if (k + 32 < K) {
      __builtin_prefetch((const void*)(pa + k + 32), 0, 1);
      __builtin_prefetch((const void*)(pw + k + 32), 0, 1);
    }
    u16x16 av;
#pragma unroll
    for (int i = 0; i < 8; ++i) { av[i] = alo[i]; av[i + 8] = ahi[i]; }
    v16bf afrag = __builtin_bit_cast(v16bf, av);
#pragma unroll
    for (int j = 0; j < NT; ++j) {
      u16x16 bv = *(const u16x16*)(pw + (size_t)j * 16 * ldw + k);
      acc[j] = __builtin_amdgcn_wmma_f32_16x16x32_bf16(
          false, afrag, false, __builtin_bit_cast(v16bf, bv),
          (short)0, acc[j], false, false);
    }
  }
  // C layout: VGPR r -> row (r + 8*ksel), col lane&15
  int crow = m0 + (ksel << 3);
#pragma unroll
  for (int j = 0; j < NT; ++j) {
    int ccol = n0 + j * 16 + r15;
#pragma unroll
    for (int r = 0; r < 8; ++r) {
      size_t off = (size_t)(crow + r) * ldc + ccol;
      float v = acc[j][r];
      if (addsrc) v += addsrc[off];
      C[off] = v;
    }
  }
}

// ---------------- depthwise causal conv (K=12) + SiLU; writes f32 + bf16 ----------------
__global__ void dwconv_silu_kernel(const float* __restrict__ xz,
                                   const float* __restrict__ conv_w,
                                   const float* __restrict__ conv_b,
                                   float* __restrict__ u_f32,
                                   unsigned short* __restrict__ u_bf16) {
  int idx = blockIdx.x * blockDim.x + threadIdx.x;
  if (idx >= NTOK * DINNER) return;
  int d   = idx % DINNER;
  int tok = idx / DINNER;
  int t   = tok % T_SZ;
  int bT  = tok - t;
  float acc = conv_b[d];
#pragma unroll
  for (int j = 0; j < DCONV; ++j) {
    int tt = t - (DCONV - 1) + j;
    if (tt >= 0)
      acc += xz[(size_t)(bT + tt) * (2 * DINNER) + d] * conv_w[d * DCONV + j];
  }
  float s = acc * fsigmoid(acc);
  u_f32[idx]  = s;
  u_bf16[idx] = f32_to_bf16_rne(s);
}

// ---------------- dt_proj (K=10) + softplus ----------------
__global__ void dt_softplus_kernel(const float* __restrict__ x_dbl,
                                   const float* __restrict__ dt_w,
                                   const float* __restrict__ dt_b,
                                   float* __restrict__ delta) {
  int idx = blockIdx.x * blockDim.x + threadIdx.x;
  if (idx >= NTOK * DINNER) return;
  int d   = idx % DINNER;
  int tok = idx / DINNER;
  const float* xr = x_dbl + (size_t)tok * XPN;
  float acc = dt_b[d];
#pragma unroll
  for (int r = 0; r < DTRANK; ++r) acc += xr[r] * dt_w[d * DTRANK + r];
  delta[idx] = (acc > 20.0f) ? acc : log1pf(__expf(acc));
}

// ---------------- selective scan ----------------
// One wave per (batch, 4 channels); lanes own 2 of 64 states per channel.
// B_t/C_t (512B) double-buffered in LDS via global_load_async_to_lds_b128,
// prefetching t+1 while t computes (ASYNCcnt-synchronized).
#define SCAN_CPW 4
__global__ void selective_scan_kernel(const float* __restrict__ delta,
                                      const float* __restrict__ u_f32,
                                      const float* __restrict__ x_dbl,
                                      const float* __restrict__ xz,
                                      const float* __restrict__ A_log,
                                      const float* __restrict__ D_skip,
                                      unsigned short* __restrict__ y_bf16) {
  __shared__ float ldsbc[4][2][2 * DSTATE];   // [wave][buf][B(64)|C(64)]
  int w    = threadIdx.x >> 5;
  int wid  = blockIdx.x * (blockDim.x >> 5) + w;
  int lane = threadIdx.x & 31;
  const int grp = DINNER / SCAN_CPW;          // 80
  if (wid >= B_SZ * grp) return;
  int b  = wid / grp;
  int d0 = (wid % grp) * SCAN_CPW;

  float a0[SCAN_CPW], a1[SCAN_CPW], dsk[SCAN_CPW], h0[SCAN_CPW], h1[SCAN_CPW];
#pragma unroll
  for (int c = 0; c < SCAN_CPW; ++c) {
    a0[c]  = -__expf(A_log[(d0 + c) * DSTATE + lane]);
    a1[c]  = -__expf(A_log[(d0 + c) * DSTATE + lane + 32]);
    dsk[c] = D_skip[d0 + c];
    h0[c] = 0.f; h1[c] = 0.f;
  }
  size_t tokBase = (size_t)b * T_SZ;
  unsigned lds_base = (unsigned)(size_t)(&ldsbc[w][0][0]);  // LDS byte offset (addr[31:0])
  // prime the pipeline: B/C block of t=0 (x_dbl cols 16..143, 16B aligned)
  {
    const char* g = (const char*)(x_dbl + tokBase * XPN + XP_BOFF) + 16 * lane;
    async_load_b128(lds_base + 16 * lane, g);
  }
  for (int t = 0; t < T_SZ; ++t) {
    size_t tok = tokBase + t;
    wait_async0();                               // buffer for t is resident
    if (t + 1 < T_SZ) {                          // prefetch t+1 into other buffer
      wait_ds0();                                // WAR guard vs. prior-step LDS reads
      const char* g = (const char*)(x_dbl + (tok + 1) * XPN + XP_BOFF) + 16 * lane;
      async_load_b128(lds_base + (((t + 1) & 1) ? 512u : 0u) + 16 * lane, g);
    }
    const float* bc = &ldsbc[w][t & 1][0];
    float bb0 = bc[lane],      bb1 = bc[lane + 32];
    float cc0 = bc[64 + lane], cc1 = bc[64 + lane + 32];

    float4 dv = *(const float4*)(delta + tok * DINNER + d0);   // broadcast loads
    float4 uv = *(const float4*)(u_f32 + tok * DINNER + d0);
    float dlt[SCAN_CPW] = {dv.x, dv.y, dv.z, dv.w};
    float ut[SCAN_CPW]  = {uv.x, uv.y, uv.z, uv.w};
    float tot[SCAN_CPW];
#pragma unroll
    for (int c = 0; c < SCAN_CPW; ++c) {
      float du = dlt[c] * ut[c];
      h0[c] = h0[c] * __expf(dlt[c] * a0[c]) + du * bb0;
      h1[c] = h1[c] * __expf(dlt[c] * a1[c]) + du * bb1;
      float p = h0[c] * cc0 + h1[c] * cc1;
#pragma unroll
      for (int m = 16; m >= 1; m >>= 1) p += __shfl_xor(p, m, 32);
      tot[c] = p;
    }
    float4 zv = *(const float4*)(xz + tok * (2 * DINNER) + DINNER + d0);
    float zz[SCAN_CPW] = {zv.x, zv.y, zv.z, zv.w};
#pragma unroll
    for (int c = 0; c < SCAN_CPW; ++c) {
      if (lane == c) {
        float yv = (tot[c] + ut[c] * dsk[c]) * (zz[c] * fsigmoid(zz[c]));
        y_bf16[tok * DINNER + d0 + c] = f32_to_bf16_rne(yv);
      }
    }
  }
}

// ---------------- head: out[tok] = dot(h[tok,:], head_w) + head_b ----------------
__global__ void head_kernel(const float* __restrict__ h,
                            const float* __restrict__ head_w,
                            const float* __restrict__ head_b,
                            float* __restrict__ out) {
  int wid  = blockIdx.x * (blockDim.x >> 5) + (threadIdx.x >> 5);
  int lane = threadIdx.x & 31;
  if (wid >= NTOK) return;
  const float* row = h + (size_t)wid * DMODEL;
  float s = 0.f;
#pragma unroll
  for (int i = 0; i < 5; ++i) s += row[lane + 32 * i] * head_w[lane + 32 * i];
#pragma unroll
  for (int m = 16; m >= 1; m >>= 1) s += __shfl_xor(s, m, 32);
  if (lane == 0) out[wid] = s + head_b[0];
}

// ---------------- host launcher ----------------
extern "C" void kernel_launch(void* const* d_in, const int* in_sizes, int n_in,
                              void* d_out, int out_size, void* d_ws, size_t ws_size,
                              hipStream_t stream) {
  (void)in_sizes; (void)n_in; (void)out_size; (void)ws_size;
  const float* x       = (const float*)d_in[0];
  const float* pre_w   = (const float*)d_in[1];
  const float* pre_b   = (const float*)d_in[2];
  const float* ln_g    = (const float*)d_in[3];
  const float* ln_b    = (const float*)d_in[4];
  const float* in_w    = (const float*)d_in[5];
  const float* conv_w  = (const float*)d_in[6];
  const float* conv_b  = (const float*)d_in[7];
  const float* xproj_w = (const float*)d_in[8];
  const float* dt_w    = (const float*)d_in[9];
  const float* dt_b    = (const float*)d_in[10];
  const float* A_log   = (const float*)d_in[11];
  const float* D_skip  = (const float*)d_in[12];
  const float* out_w   = (const float*)d_in[13];
  const float* head_w  = (const float*)d_in[14];
  const float* head_b  = (const float*)d_in[15];
  float* out = (float*)d_out;

  char* ws = (char*)d_ws;
  size_t off = 0;
  auto alloc = [&](size_t bytes) -> char* {
    char* p = ws + off;
    off += (bytes + 255) & ~(size_t)255;
    return p;
  };
  unsigned short* wb_in  = (unsigned short*)alloc((size_t)INW_CNT * 2);
  unsigned short* wb_xp  = (unsigned short*)alloc((size_t)XPW_CNT * 2);
  unsigned short* wb_out = (unsigned short*)alloc((size_t)OUTW_CNT * 2);
  float*          hbuf   = (float*)alloc((size_t)NTOK * DMODEL * 4);
  unsigned short* ln_bf  = (unsigned short*)alloc((size_t)NTOK * DMODEL * 2);
  float*          xz     = (float*)alloc((size_t)NTOK * 2 * DINNER * 4);
  float*          u_f    = (float*)alloc((size_t)NTOK * DINNER * 4);
  unsigned short* u_bf   = (unsigned short*)alloc((size_t)NTOK * DINNER * 2);
  float*          x_dbl  = (float*)alloc((size_t)NTOK * XPN * 4);
  float*          delta  = (float*)alloc((size_t)NTOK * DINNER * 4);
  unsigned short* y_bf   = (unsigned short*)alloc((size_t)NTOK * DINNER * 2);

  const int thr = 256;

  {
    int total = INW_CNT + XPW_CNT + OUTW_CNT;
    convert_weights_kernel<<<(total + thr - 1) / thr, thr, 0, stream>>>(
        in_w, xproj_w, out_w, wb_in, wb_xp, wb_out);
  }
  stem_gelu_kernel<<<(NTOK * DMODEL + thr - 1) / thr, thr, 0, stream>>>(
      x, pre_w, pre_b, hbuf);

  for (int blk = 0; blk < NBLK; ++blk) {
    layernorm_bf16_kernel<<<(NTOK + 3) / 4, 128, 0, stream>>>(
        hbuf, ln_g + blk * DMODEL, ln_b + blk * DMODEL, ln_bf);

    {  // in_proj: M=NTOK, N=640, K=160; 4 tiles/wave
      int waves = (NTOK / 16) * ((2 * DINNER) / 64);
      gemm_bf16_wmma_kernel<4><<<(waves + 3) / 4, 128, 0, stream>>>(
          ln_bf, DMODEL,
          wb_in + (size_t)blk * 2 * DINNER * DMODEL, DMODEL,
          nullptr, xz, 2 * DINNER, NTOK, 2 * DINNER, DMODEL);
    }

    dwconv_silu_kernel<<<(NTOK * DINNER + thr - 1) / thr, thr, 0, stream>>>(
        xz, conv_w + blk * DINNER * DCONV, conv_b + blk * DINNER, u_f, u_bf);

    {  // x_proj: M=NTOK, N=144 (padded/remapped), K=320; 3 tiles/wave
      int waves = (NTOK / 16) * (XPN / 48);
      gemm_bf16_wmma_kernel<3><<<(waves + 3) / 4, 128, 0, stream>>>(
          u_bf, DINNER,
          wb_xp + (size_t)blk * XPN * DINNER, DINNER,
          nullptr, x_dbl, XPN, NTOK, XPN, DINNER);
    }

    dt_softplus_kernel<<<(NTOK * DINNER + thr - 1) / thr, thr, 0, stream>>>(
        x_dbl, dt_w + blk * DINNER * DTRANK, dt_b + blk * DINNER, delta);

    selective_scan_kernel<<<(B_SZ * (DINNER / SCAN_CPW) + 3) / 4, 128, 0, stream>>>(
        delta, u_f, x_dbl, xz,
        A_log + (size_t)blk * DINNER * DSTATE, D_skip + blk * DINNER, y_bf);

    {  // out_proj + residual (fused): M=NTOK, N=160, K=320; 5 tiles/wave
      int waves = (NTOK / 16) * (DMODEL / 80);
      gemm_bf16_wmma_kernel<5><<<(waves + 3) / 4, 128, 0, stream>>>(
          y_bf, DINNER,
          wb_out + (size_t)blk * DMODEL * DINNER, DINNER,
          hbuf, hbuf, DMODEL, NTOK, DMODEL, DINNER);
    }
  }

  head_kernel<<<(NTOK + 3) / 4, 128, 0, stream>>>(hbuf, head_w, head_b, out);
}
